// MultiLinearAttention_35562329211841
// MI455X (gfx1250) — matrix operand order repaired and checked by reading, not compiled, not run on hardware
//
#include <hip/hip_runtime.h>

// ---------------------------------------------------------------------------
// Linear attention (elu+1 feature map) for MI455X / gfx1250, wave32 + WMMA.
//   kv   = K'^T V          (64x64xS GEMM per head)  -> v_wmma_f32_16x16x32_f16
//   out  = (Q' kv) / (Q'.ksum + eps)                -> v_wmma_f32_16x16x32_f16
// f16 operands staged through LDS (values are O(1); f32 accumulation in WMMA).
// HBM-bound problem: ~256MB streaming @ 23.3 TB/s ~= 11 us. Design:
//  - batched unconditional NT global loads (full MLP, clause-friendly)
//  - branchless feature map, multiplicative masking (K' only; valid^2==valid)
//  - wide LDS ops only (b128/b64), 16B-aligned padded tiles
//  - double-buffered LDS in the kv pass: 1 barrier/step, next step's global
//    loads overlap current step's WMMAs + conversions.
// ---------------------------------------------------------------------------

typedef __attribute__((ext_vector_type(16))) _Float16 v16h;
typedef __attribute__((ext_vector_type(8)))  _Float16 v8h;
typedef __attribute__((ext_vector_type(4)))  _Float16 v4h;
typedef __attribute__((ext_vector_type(8)))  float    v8f;
typedef __attribute__((ext_vector_type(4)))  float    v4f;
typedef __attribute__((ext_vector_type(4)))  int      v4i;

#define WMMA_F16(a, b, c) \
  __builtin_amdgcn_wmma_f32_16x16x32_f16(false, (a), false, (b), (short)0, (c), false, false)

#define SHUF16(lo, hi) \
  __builtin_shufflevector((lo), (hi), 0,1,2,3,4,5,6,7,8,9,10,11,12,13,14,15)

constexpr int Bc = 4, Hc = 16, Sc = 4096, Dc = 64;
constexpr int BH = Bc * Hc;              // 64 (b,h) pairs
constexpr int NCH = 4;                   // S-chunks per head for kv pass
constexpr int CHUNK = Sc / NCH;          // 1024 rows per chunk
constexpr int SB = 64;                   // s-rows staged per step (2 WMMAs/wave)
constexpr int NSTEP = CHUNK / SB;        // 16 pipelined steps
constexpr float EPS = 1e-6f;

// Workspace layout (in floats)
constexpr size_t KVP_OFF = 0;                                   // kv partials  [BH][NCH][64*64] ([d][f])
constexpr size_t KSP_OFF = KVP_OFF + (size_t)BH * NCH * 4096;   // ksum partials[BH][NCH][64]
constexpr size_t KVF_OFF = KSP_OFF + (size_t)BH * NCH * 64;     // kv final     [BH][64*64] TRANSPOSED ([f][d])
constexpr size_t KSF_OFF = KVF_OFF + (size_t)BH * 4096;         // ksum final   [BH][64]

__device__ __forceinline__ float fmap(float x) {
  // elu(x) + 1 == (x > 0) ? x + 1 : exp(x)   (branchless: v_cndmask + v_exp)
  return x > 0.0f ? x + 1.0f : __expf(x);
}

// ---------------------------------------------------------------------------
// Kernel 1: per-(b,h,chunk) partial kv = K'^T V and ksum = sum_s K'[s,:]
// 512 threads = 16 waves; wave w owns the (dtile = w>>2, ftile = w&3) 16x16
// tile of the 64x64 kv accumulator. Staging: thread owns column d = tid&63
// and 8 consecutive s-rows -> coalesced b32 global loads (lanes sweep d),
// one packed b128 LDS store per matrix into the transposed tile.
// Double-buffered LDS: one barrier per 64-row step.
// ---------------------------------------------------------------------------
__global__ __launch_bounds__(512) void kv_kernel(const float* __restrict__ K,
                                                 const float* __restrict__ V,
                                                 const int* __restrict__ mask,
                                                 float* __restrict__ ws) {
  const int bh   = blockIdx.x >> 2;         // NCH == 4
  const int ch   = blockIdx.x & 3;
  const int b    = bh >> 4;                 // / H
  const int tid  = threadIdx.x;
  const int lane = tid & 31;
  const int wave = tid >> 5;
  const int dtile = wave >> 2;
  const int ftile = wave & 3;

  // Row stride 72 halves (144B, multiple of 16B) -> aligned b128 accesses.
  __shared__ __align__(16) _Float16 kT[2][64][72];   // kT[buf][d][srow] = K'[s][d]
  __shared__ __align__(16) _Float16 vT[2][64][72];   // vT[buf][f][srow] = V[s][f]
  __shared__ float red[512];

  const float* kg = K + (size_t)bh * Sc * Dc;
  const float* vg = V + (size_t)bh * Sc * Dc;
  const int*   mb = mask + (size_t)b * Sc;
  const int    s0 = ch * CHUNK;

  v8f   acc  = {};
  float ksum = 0.0f;

  // Staging assignment: fixed column d, 8 consecutive rows starting at srow0.
  const int d     = tid & 63;
  const int srow0 = (tid >> 6) * 8;         // 0,8,...,56

  // 16-bit A 16x32 fragment: lanes 0-15 rows M, elems 0..7 -> K kb..kb+7,
  // elems 8..15 -> K kb+16..kb+23, kb = (lane>=16)*8.
  const int dA  = dtile * 16 + (lane & 15);
  const int kbA = (lane >> 4) * 8;
  // 16-bit B 32x16 fragment: lane holds col n, elems 0..7 -> K kb..kb+7,
  // elems 8..15 -> K kb+16..kb+23, kb = (lane>=16)*8.
  const int fB  = ftile * 16 + (lane & 15);
  const int kbB = (lane >> 4) * 8;

  // In-flight register tile for the next step.
  v4i   m0, m1;
  float kr[8], vr[8];

  auto load_step = [&](int step) {
    const int sbase = s0 + step * SB + srow0;
    m0 = *(const v4i*)(mb + sbase);
    m1 = *(const v4i*)(mb + sbase + 4);
#pragma unroll
    for (int j = 0; j < 8; ++j)
      kr[j] = __builtin_nontemporal_load(kg + (size_t)(sbase + j) * Dc + d);
#pragma unroll
    for (int j = 0; j < 8; ++j)
      vr[j] = __builtin_nontemporal_load(vg + (size_t)(sbase + j) * Dc + d);
  };

  auto stage_step = [&](int buf) {
    const int m[8] = {m0[0], m0[1], m0[2], m0[3], m1[0], m1[1], m1[2], m1[3]};
    v8h kh, vh;
#pragma unroll
    for (int j = 0; j < 8; ++j) {
      const float valid = (m[j] == 0) ? 1.0f : 0.0f;   // single v_cndmask
      const float kf = fmap(kr[j]) * valid;            // fmap always live
      ksum += kf;
      kh[j] = (_Float16)kf;
      vh[j] = (_Float16)vr[j];       // V needs no mask: (valid*k')(x)(valid*v) == (valid*k')(x)v
    }
    *(v8h*)&kT[buf][d][srow0] = kh;  // one ds_store_b128
    *(v8h*)&vT[buf][d][srow0] = vh;  // one ds_store_b128
  };

  auto do_wmma = [&](int buf) {
#pragma unroll
    for (int c = 0; c < 2; ++c) {
      const v8h alo = *(const v8h*)&kT[buf][dA][c * 32 + kbA];
      const v8h ahi = *(const v8h*)&kT[buf][dA][c * 32 + kbA + 16];
      const v8h blo = *(const v8h*)&vT[buf][fB][c * 32 + kbB];
      const v8h bhi = *(const v8h*)&vT[buf][fB][c * 32 + kbB + 16];
      acc = WMMA_F16(SHUF16(alo, ahi), SHUF16(blo, bhi), acc);
    }
  };

  // Pipelined main loop: 1 barrier per step; step i+1's global loads overlap
  // step i's WMMAs + conversions.
  load_step(0);
  stage_step(0);
  for (int i = 0; i < NSTEP - 1; ++i) {
    load_step(i + 1);                 // issue next tile's NT loads early
    __syncthreads();                  // buf[i&1] fully staged for all waves
    do_wmma(i & 1);
    stage_step((i + 1) & 1);          // other buffer; guarded by the barrier
  }
  __syncthreads();
  do_wmma((NSTEP - 1) & 1);

  // ksum reduction: 8 partials per column d (threads d, d+64, ..., d+448).
  red[tid] = ksum;
  __syncthreads();
  if (tid < 64) {
    float t = 0.0f;
#pragma unroll
    for (int j = 0; j < 8; ++j) t += red[tid + 64 * j];
    ws[KSP_OFF + ((size_t)bh * NCH + ch) * 64 + tid] = t;
  }

  // Write kv partial tile ([d][f] layout).
  // C/D layout: VGPR i -> M = i + (lane>=16)*8, N = lane&15.
  float* kvp = ws + KVP_OFF + ((size_t)bh * NCH + ch) * 4096;
#pragma unroll
  for (int i = 0; i < 8; ++i) {
    const int mrow = dtile * 16 + i + ((lane >> 4) << 3);
    const int ncol = ftile * 16 + (lane & 15);
    kvp[mrow * 64 + ncol] = acc[i];
  }
}

// ---------------------------------------------------------------------------
// Kernel 2: deterministic reduction of the NCH chunk partials.
// kv is written TRANSPOSED ([f][d]) so the out-pass can stage it with
// contiguous b128 loads along d.
// ---------------------------------------------------------------------------
__global__ void reduce_kernel(float* __restrict__ ws) {
  const int e = blockIdx.x * blockDim.x + threadIdx.x;
  const int NKV = BH * 4096;
  if (e < NKV) {
    const int bh = e >> 12, r = e & 4095;       // r = d*64 + f
    float t = 0.0f;
#pragma unroll
    for (int c = 0; c < NCH; ++c) t += ws[KVP_OFF + ((size_t)bh * NCH + c) * 4096 + r];
    ws[KVF_OFF + (size_t)bh * 4096 + (r & 63) * 64 + (r >> 6)] = t;   // [f][d]
  } else if (e - NKV < BH * 64) {
    const int e2 = e - NKV;
    const int bh = e2 >> 6, r = e2 & 63;
    float t = 0.0f;
#pragma unroll
    for (int c = 0; c < NCH; ++c) t += ws[KSP_OFF + ((size_t)bh * NCH + c) * 64 + r];
    ws[KSF_OFF + e2] = t;
  }
}

// ---------------------------------------------------------------------------
// Kernel 3: out = (Q' kv) / (Q'.ksum + eps).  One block per (b,h, 64-row tile).
// 16 waves: wave w owns (ttile = w>>2, ftile = w&3); K-dim (d=64) = 2 WMMAs.
// Staging via b128 loads + v4h b64 LDS stores; z computed in f32 during
// staging (4-wide partial dot per thread, 16 partials per row reduced in LDS).
// ---------------------------------------------------------------------------
__global__ __launch_bounds__(512) void out_kernel(const float* __restrict__ Q,
                                                  const float* __restrict__ ws,
                                                  float* __restrict__ Out) {
  const int bh   = blockIdx.x >> 6;
  const int rb   = blockIdx.x & 63;        // 64-row tile index within S
  const int tid  = threadIdx.x;
  const int lane = tid & 31;
  const int wave = tid >> 5;
  const int ttile = wave >> 2;
  const int ftile = wave & 3;

  __shared__ __align__(16) _Float16 qs[64][72];    // qs[t][d]  = Q'[t][d]
  __shared__ __align__(16) _Float16 kvT[64][72];   // kvT[f][d] = kv[d][f]
  __shared__ float ksums[64];
  __shared__ float zpart[64][16];
  __shared__ float zf[64];

  const float* qg   = Q + (size_t)bh * Sc * Dc + (size_t)rb * 64 * Dc;
  const float* kvfT = ws + KVF_OFF + (size_t)bh * 4096;   // [f][d]

  if (tid < 64) ksums[tid] = ws[KSF_OFF + (size_t)bh * 64 + tid];
  __syncthreads();

  const int d4 = (tid & 15) * 4;
  const int r0 = tid >> 4;                 // 0..31 (+32 on second pass)
#pragma unroll
  for (int p = 0; p < 2; ++p) {
    const int r = r0 + p * 32;
    // q: streaming (NT b128); kv table: cache-resident (RT b128, reused 64x).
    const v4f qv  = __builtin_nontemporal_load((const v4f*)(qg + (size_t)r * Dc + d4));
    const v4f kv4 = *(const v4f*)(kvfT + (size_t)r * 64 + d4);

    const float q0 = fmap(qv[0]), q1 = fmap(qv[1]), q2 = fmap(qv[2]), q3 = fmap(qv[3]);
    zpart[r][tid & 15] = q0 * ksums[d4] + q1 * ksums[d4 + 1] +
                         q2 * ksums[d4 + 2] + q3 * ksums[d4 + 3];

    v4h qh;  qh[0] = (_Float16)q0;     qh[1] = (_Float16)q1;
             qh[2] = (_Float16)q2;     qh[3] = (_Float16)q3;
    v4h kh;  kh[0] = (_Float16)kv4[0]; kh[1] = (_Float16)kv4[1];
             kh[2] = (_Float16)kv4[2]; kh[3] = (_Float16)kv4[3];
    *(v4h*)&qs[r][d4]  = qh;          // ds_store_b64
    *(v4h*)&kvT[r][d4] = kh;          // ds_store_b64
  }
  __syncthreads();

  if (tid < 64) {
    float zz = 0.0f;
#pragma unroll
    for (int j = 0; j < 16; ++j) zz += zpart[tid][j];
    zf[tid] = zz + EPS;
  }
  __syncthreads();

  const int tA  = ttile * 16 + (lane & 15);
  const int fB  = ftile * 16 + (lane & 15);
  v8f acc = {};
#pragma unroll
  for (int c = 0; c < 2; ++c) {            // K = d in two 32-chunks
    const int kbA = c * 32 + ((lane >> 4) << 3);
    const int kbB = c * 32 + ((lane >> 4) << 3);
    const v8h alo = *(const v8h*)&qs[tA][kbA];
    const v8h ahi = *(const v8h*)&qs[tA][kbA + 16];
    const v8h blo = *(const v8h*)&kvT[fB][kbB];
    const v8h bhi = *(const v8h*)&kvT[fB][kbB + 16];
    acc = WMMA_F16(SHUF16(alo, ahi), SHUF16(blo, bhi), acc);
  }

  float* og = Out + (size_t)bh * Sc * Dc + (size_t)rb * 64 * Dc;
#pragma unroll
  for (int i = 0; i < 8; ++i) {
    const int mrow = ttile * 16 + i + ((lane >> 4) << 3);
    const int ncol = ftile * 16 + (lane & 15);
    __builtin_nontemporal_store(acc[i] / zf[mrow], og + (size_t)mrow * 64 + ncol);
  }
}

// ---------------------------------------------------------------------------
extern "C" void kernel_launch(void* const* d_in, const int* in_sizes, int n_in,
                              void* d_out, int out_size, void* d_ws, size_t ws_size,
                              hipStream_t stream) {
  const float* Q    = (const float*)d_in[0];   // query  [4,16,4096,64] f32
  const float* K    = (const float*)d_in[1];   // key    [4,16,4096,64] f32
  const float* V    = (const float*)d_in[2];   // value  [4,16,4096,64] f32
  const int*   mask = (const int*)d_in[3];     // key_padding_mask [4,4096] (nonzero = padded)
  float* out = (float*)d_out;
  float* ws  = (float*)d_ws;                   // needs ~5.1 MB

  kv_kernel<<<BH * NCH, 512, 0, stream>>>(K, V, mask, ws);

  const int ntot = BH * 4096 + BH * 64;
  reduce_kernel<<<(ntot + 255) / 256, 256, 0, stream>>>(ws);

  out_kernel<<<BH * 64, 512, 0, stream>>>(Q, ws, out);
}
